// FlashAttention_71416716198218
// MI455X (gfx1250) — compile-verified, matrix operand order checked by reading, not run on hardware
//
#include <hip/hip_runtime.h>
#include <hip/hip_bf16.h>

typedef __attribute__((ext_vector_type(16))) _Float16 v16h;
typedef __attribute__((ext_vector_type(8)))  _Float16 v8h;
typedef __attribute__((ext_vector_type(8)))  float    v8f;

#define DIMC  512
#define NHEAD 8
#define HDIM  64
#define BATCH 4
#define SEQ   2048
#define MTOT  (BATCH*SEQ)          // 8192
#define SCALE 0.015625f            // 64^-0.5 * 0.125

union Frag16 { v16h f; v8h h[2]; };

// Load a 16x32 f16 WMMA fragment from a row-major matrix (A from MxK, or B as
// B^T from NxK).  ISA layout (CDNA5 ISA, 16-bit A 16x32):
//   lanes 0-15  : row = lane,    V0..3 = K 0..7,   V4..7 = K 16..23
//   lanes 16-31 : row = lane-16, V0..3 = K 8..15,  V4..7 = K 24..31
__device__ __forceinline__ v16h load_frag(const _Float16* base, int ld, int lane) {
  const int row = lane & 15;
  const int ko  = (lane >> 4) << 3;
  const _Float16* p = base + row * ld + ko;
  Frag16 u;
  u.h[0] = *(const v8h*)(p);
  u.h[1] = *(const v8h*)(p + 16);
  return u.f;
}

__device__ __forceinline__ v8f wmma_f16(v16h a, v16h b, v8f c) {
  return __builtin_amdgcn_wmma_f32_16x16x32_f16(false, a, false, b, (short)0, c,
                                                false, false);
}

// CDNA5 async global->LDS copy (GLOBAL_LOAD_ASYNC_TO_LDS_B128, ASYNCcnt).
// 16 bytes per active lane; LDS dest address = low 32 bits of generic pointer.
__device__ __forceinline__ void async_load_b128(void* lds_dst, const void* gsrc) {
  unsigned l = (unsigned)(unsigned long long)lds_dst;
  asm volatile("global_load_async_to_lds_b128 %0, %1, off"
               :: "v"(l), "v"(gsrc)
               : "memory");
}
__device__ __forceinline__ void wait_async0() {
  asm volatile("s_wait_asynccnt 0x0" ::: "memory");
}
// Compiler-only ordering fence for wave-synchronous LDS reuse (same-wave DS
// ops are in-order in hardware).
__device__ __forceinline__ void lds_fence() {
  asm volatile("" ::: "memory");
}

// ---------------------------------------------------------------- conversions
__global__ void f32_to_f16_kernel(const float* __restrict__ in,
                                  _Float16* __restrict__ out, int n) {
  int i = blockIdx.x * blockDim.x + threadIdx.x;
  if (i < n) out[i] = (_Float16)in[i];
}

// in[rows][cols] (f32) -> out[cols][rows] (f16)
__global__ void transpose_to_f16_kernel(const float* __restrict__ in,
                                        _Float16* __restrict__ out,
                                        int rows, int cols) {
  int i = blockIdx.x * blockDim.x + threadIdx.x;
  if (i < rows * cols) {
    int c = i % cols, r = i / cols;
    out[(size_t)c * rows + r] = (_Float16)in[(size_t)r * cols + c];
  }
}

// ------------------------------------------------------------------ QKV GEMM
// C[8192,1536] = A[8192,512] x W + bias, scattered into Q,K ([B,H,N,64] f16)
// and V transposed ([B,H,64,N] f16).  Weight tile double-buffered in LDS via
// async global->LDS copies.
__global__ __launch_bounds__(256) void gemm_qkv_kernel(
    const _Float16* __restrict__ A,   // [8192,512] row-major
    const _Float16* __restrict__ Bt,  // [1536,512] row-major (W^T)
    const float* __restrict__ bias,   // [1536]
    _Float16* __restrict__ Qh, _Float16* __restrict__ Kh,
    _Float16* __restrict__ Vt) {
  __shared__ _Float16 Bs[2][64 * 128];
  const int tid = threadIdx.x, lane = tid & 31, w = tid >> 5;
  const int half = lane >> 4, colb = lane & 15;
  const int m0 = blockIdx.y * 128 + w * 16;
  const int n0 = blockIdx.x * 64;
  const _Float16* Arow = A + (size_t)m0 * DIMC;

  auto issue_chunk = [&](int buf, int kc) {
#pragma unroll
    for (int it = 0; it < 4; ++it) {
      int idx = tid + it * 256;        // 0..1023
      int r = idx >> 4;                // 0..63
      int c8 = (idx & 15) << 3;        // 0..120
      async_load_b128(&Bs[buf][r * 128 + c8],
                      &Bt[(size_t)(n0 + r) * DIMC + kc + c8]);
    }
  };

  v8f acc[4] = {};
  issue_chunk(0, 0);
  wait_async0();
  __syncthreads();

  for (int kci = 0; kci < 4; ++kci) {
    const int kc = kci * 128, cur = kci & 1;
    if (kci + 1 < 4) issue_chunk(1 - cur, kc + 128);
#pragma unroll
    for (int ks = 0; ks < 128; ks += 32) {
      v16h a  = load_frag(Arow + kc + ks, DIMC, lane);
      v16h b0 = load_frag(&Bs[cur][0 * 2048 + ks], 128, lane);
      v16h b1 = load_frag(&Bs[cur][1 * 2048 + ks], 128, lane);
      v16h b2 = load_frag(&Bs[cur][2 * 2048 + ks], 128, lane);
      v16h b3 = load_frag(&Bs[cur][3 * 2048 + ks], 128, lane);
      acc[0] = wmma_f16(a, b0, acc[0]);
      acc[1] = wmma_f16(a, b1, acc[1]);
      acc[2] = wmma_f16(a, b2, acc[2]);
      acc[3] = wmma_f16(a, b3, acc[3]);
    }
    if (kci + 1 < 4) {
      wait_async0();        // next chunk landed (overlapped with compute)
      __syncthreads();      // publish across waves / protect buffer reuse
    }
  }

  // Scatter stores.  Each 16-col fragment stays inside one of {Q,K,V} and one
  // head (all boundaries are multiples of 64).
#pragma unroll
  for (int f = 0; f < 4; ++f) {
    int col = n0 + f * 16 + colb;
    float bv = bias[col];
    int which = col >> 9;            // 0=Q 1=K 2=V
    int cc = col & 511;
    int hh = cc >> 6;
    int d = cc & 63;
    if (which == 2) {
      // V transposed: C-fragment rows are consecutive n at fixed d -> b128
      int b = m0 >> 11;
      int nb = (m0 & 2047) + half * 8;
      v8h pk;
#pragma unroll
      for (int r = 0; r < 8; ++r) pk[r] = (_Float16)(acc[f][r] + bv);
      *(v8h*)&Vt[((size_t)(b * NHEAD + hh) * HDIM + d) * SEQ + nb] = pk;
    } else {
      _Float16* dst = (which == 0) ? Qh : Kh;
#pragma unroll
      for (int r = 0; r < 8; ++r) {
        int m = m0 + half * 8 + r;
        int b = m >> 11, n = m & 2047;
        dst[((size_t)(b * NHEAD + hh) * SEQ + n) * HDIM + d] =
            (_Float16)(acc[f][r] + bv);
      }
    }
  }
}

// ------------------------------------------------------------ flash attention
// One block = (b, h, 64 Q rows).  4 waves, each owns a 16x64 output strip.
// K and V^T tiles double-buffered in LDS via async global->LDS copies.
__global__ __launch_bounds__(128) void flash_attn_kernel(
    const _Float16* __restrict__ Qh, const _Float16* __restrict__ Kh,
    const _Float16* __restrict__ Vt, const int* __restrict__ mask,
    const float* __restrict__ temp, _Float16* __restrict__ Oh) {
  __shared__ _Float16 Ks[2][64 * 64];     // K tile, row-major [n][d]
  __shared__ _Float16 Vts[2][64 * 64];    // V^T tile, row-major [d][n]
  __shared__ _Float16 Ps[4][16 * 64];     // per-wave P staging (C -> A layout)

  const int tid = threadIdx.x;
  const int lane = tid & 31, w = tid >> 5;
  const int half = lane >> 4, colb = lane & 15;
  const int qt = blockIdx.x, h = blockIdx.y, b = blockIdx.z;
  const int bh = b * NHEAD + h;
  const _Float16* Qg = Qh + (size_t)bh * SEQ * HDIM;
  const _Float16* Kg = Kh + (size_t)bh * SEQ * HDIM;
  const _Float16* Vg = Vt + (size_t)bh * HDIM * SEQ;
  const int qrow0 = qt * 64 + w * 16;
  const float sc = SCALE * temp[h];
  const int NT = SEQ / 64;

  auto issue_tile = [&](int buf, int jn) {
#pragma unroll
    for (int it = 0; it < 4; ++it) {
      int idx = tid + it * 128;        // 0..511
      int r = idx >> 3;                // 0..63
      int c8 = (idx & 7) << 3;         // 0..56
      async_load_b128(&Ks[buf][r * 64 + c8],
                      &Kg[(size_t)(jn + r) * HDIM + c8]);
      async_load_b128(&Vts[buf][r * 64 + c8],
                      &Vg[(size_t)r * SEQ + jn + c8]);
    }
  };

  v16h aq0 = load_frag(Qg + (size_t)qrow0 * HDIM + 0, HDIM, lane);
  v16h aq1 = load_frag(Qg + (size_t)qrow0 * HDIM + 32, HDIM, lane);

  v8f accO[4] = {};
  float mr[8], lr[8];
#pragma unroll
  for (int r = 0; r < 8; ++r) { mr[r] = -1e30f; lr[r] = 0.f; }

  issue_tile(0, 0);
  wait_async0();
  __syncthreads();

  for (int jt = 0; jt < NT; ++jt) {
    const int jn = jt * 64;
    const int cur = jt & 1;
    if (jt + 1 < NT) issue_tile(1 - cur, jn + 64);   // overlap with compute

    // S = Q K^T   (B^T of (K^T) is K itself, row-major [n][d] in LDS)
    v8f s[4];
#pragma unroll
    for (int cb = 0; cb < 4; ++cb) {
      v8f a = {};
      v16h bk0 = load_frag(&Ks[cur][cb * 16 * 64 + 0], 64, lane);
      v16h bk1 = load_frag(&Ks[cur][cb * 16 * 64 + 32], 64, lane);
      a = wmma_f16(aq0, bk0, a);
      a = wmma_f16(aq1, bk1, a);
      int col = jn + cb * 16 + colb;   // one column per lane per fragment
      int mv = mask[b * SEQ + col];
#pragma unroll
      for (int r = 0; r < 8; ++r) {
        float v = a[r] * sc;
        a[r] = mv ? v : -1e9f;
      }
      s[cb] = a;
    }

    // online softmax: row reductions across the 16-lane half via shfl_xor
    float mnew[8], alpha[8], rsum[8];
#pragma unroll
    for (int r = 0; r < 8; ++r) {
      float v = fmaxf(fmaxf(s[0][r], s[1][r]), fmaxf(s[2][r], s[3][r]));
#pragma unroll
      for (int xm = 1; xm <= 8; xm <<= 1) v = fmaxf(v, __shfl_xor(v, xm, 32));
      mnew[r] = fmaxf(mr[r], v);
      alpha[r] = __expf(mr[r] - mnew[r]);
      rsum[r] = 0.f;
    }
#pragma unroll
    for (int cb = 0; cb < 4; ++cb)
#pragma unroll
      for (int r = 0; r < 8; ++r) {
        float p = __expf(s[cb][r] - mnew[r]);
        s[cb][r] = p;
        rsum[r] += p;
      }
#pragma unroll
    for (int r = 0; r < 8; ++r) {
      float v = rsum[r];
#pragma unroll
      for (int xm = 1; xm <= 8; xm <<= 1) v += __shfl_xor(v, xm, 32);
      lr[r] = lr[r] * alpha[r] + v;
      mr[r] = mnew[r];
    }
#pragma unroll
    for (int f = 0; f < 4; ++f)
#pragma unroll
      for (int r = 0; r < 8; ++r) accO[f][r] *= alpha[r];

    // P (C layout) -> per-wave LDS -> A layout.  Same-wave DS ops are
    // in-order; only a compiler fence is needed (no block barrier).
#pragma unroll
    for (int cb = 0; cb < 4; ++cb)
#pragma unroll
      for (int r = 0; r < 8; ++r)
        Ps[w][(r + half * 8) * 64 + cb * 16 + colb] = (_Float16)s[cb][r];
    lds_fence();

    // O += P x V   (B^T of V is V^T, row-major [d][n] in LDS)
#pragma unroll
    for (int kk = 0; kk < 2; ++kk) {
      v16h ap = load_frag(&Ps[w][kk * 32], 64, lane);
      v16h bv0 = load_frag(&Vts[cur][0 * 16 * 64 + kk * 32], 64, lane);
      v16h bv1 = load_frag(&Vts[cur][1 * 16 * 64 + kk * 32], 64, lane);
      v16h bv2 = load_frag(&Vts[cur][2 * 16 * 64 + kk * 32], 64, lane);
      v16h bv3 = load_frag(&Vts[cur][3 * 16 * 64 + kk * 32], 64, lane);
      accO[0] = wmma_f16(ap, bv0, accO[0]);
      accO[1] = wmma_f16(ap, bv1, accO[1]);
      accO[2] = wmma_f16(ap, bv2, accO[2]);
      accO[3] = wmma_f16(ap, bv3, accO[3]);
    }

    if (jt + 1 < NT) wait_async0();    // next K/V tile landed
    __syncthreads();                   // publish + protect buffer reuse
  }

  // epilogue: normalize and store to [B,N,H*64] f16 for the proj GEMM
#pragma unroll
  for (int db = 0; db < 4; ++db) {
    int d = h * HDIM + db * 16 + colb;
#pragma unroll
    for (int r = 0; r < 8; ++r) {
      int n = qrow0 + r + half * 8;
      float o = accO[db][r] / lr[r];
      Oh[((size_t)(b * SEQ + n)) * DIMC + d] = (_Float16)o;
    }
  }
}

// ----------------------------------------------------------------- proj GEMM
__global__ __launch_bounds__(256) void gemm_proj_kernel(
    const _Float16* __restrict__ A,   // [8192,512]
    const _Float16* __restrict__ Bt,  // [512,512] (W^T)
    const float* __restrict__ bias,   // [512]
    float* __restrict__ out) {        // [8192,512] f32
  __shared__ _Float16 Bs[2][64 * 128];
  const int tid = threadIdx.x, lane = tid & 31, w = tid >> 5;
  const int half = lane >> 4, colb = lane & 15;
  const int m0 = blockIdx.y * 128 + w * 16;
  const int n0 = blockIdx.x * 64;
  const _Float16* Arow = A + (size_t)m0 * DIMC;

  auto issue_chunk = [&](int buf, int kc) {
#pragma unroll
    for (int it = 0; it < 4; ++it) {
      int idx = tid + it * 256;
      int r = idx >> 4;
      int c8 = (idx & 15) << 3;
      async_load_b128(&Bs[buf][r * 128 + c8],
                      &Bt[(size_t)(n0 + r) * DIMC + kc + c8]);
    }
  };

  v8f acc[4] = {};
  issue_chunk(0, 0);
  wait_async0();
  __syncthreads();

  for (int kci = 0; kci < 4; ++kci) {
    const int kc = kci * 128, cur = kci & 1;
    if (kci + 1 < 4) issue_chunk(1 - cur, kc + 128);
#pragma unroll
    for (int ks = 0; ks < 128; ks += 32) {
      v16h a  = load_frag(Arow + kc + ks, DIMC, lane);
      v16h b0 = load_frag(&Bs[cur][0 * 2048 + ks], 128, lane);
      v16h b1 = load_frag(&Bs[cur][1 * 2048 + ks], 128, lane);
      v16h b2 = load_frag(&Bs[cur][2 * 2048 + ks], 128, lane);
      v16h b3 = load_frag(&Bs[cur][3 * 2048 + ks], 128, lane);
      acc[0] = wmma_f16(a, b0, acc[0]);
      acc[1] = wmma_f16(a, b1, acc[1]);
      acc[2] = wmma_f16(a, b2, acc[2]);
      acc[3] = wmma_f16(a, b3, acc[3]);
    }
    if (kci + 1 < 4) {
      wait_async0();
      __syncthreads();
    }
  }
#pragma unroll
  for (int f = 0; f < 4; ++f) {
    int col = n0 + f * 16 + colb;
    float bv = bias[col];
#pragma unroll
    for (int r = 0; r < 8; ++r) {
      int m = m0 + half * 8 + r;
      out[(size_t)m * DIMC + col] = acc[f][r] + bv;
    }
  }
}

// -------------------------------------------------------------------- launch
extern "C" void kernel_launch(void* const* d_in, const int* in_sizes, int n_in,
                              void* d_out, int out_size, void* d_ws,
                              size_t ws_size, hipStream_t stream) {
  const float* x     = (const float*)d_in[0];
  const int*   mask  = (const int*)d_in[1];
  const float* Wqkv  = (const float*)d_in[2];
  const float* bqkv  = (const float*)d_in[3];
  const float* Wproj = (const float*)d_in[4];
  const float* bproj = (const float*)d_in[5];
  const float* temp  = (const float*)d_in[6];
  float* out = (float*)d_out;

  char* ws = (char*)d_ws;
  size_t off = 0;
  auto carve = [&](size_t bytes) -> char* {
    char* p = ws + off;
    off += (bytes + 255) & ~(size_t)255;
    return p;
  };
  _Float16* xh  = (_Float16*)carve((size_t)MTOT * DIMC * 2);      // 8 MB
  _Float16* Wt1 = (_Float16*)carve((size_t)3 * DIMC * DIMC * 2);  // 1.5 MB
  _Float16* Wt2 = (_Float16*)carve((size_t)DIMC * DIMC * 2);      // 0.5 MB
  _Float16* Qh  = (_Float16*)carve((size_t)MTOT * DIMC * 2);
  _Float16* Kh  = (_Float16*)carve((size_t)MTOT * DIMC * 2);
  _Float16* Vt  = (_Float16*)carve((size_t)MTOT * DIMC * 2);
  _Float16* Oh  = (_Float16*)carve((size_t)MTOT * DIMC * 2);

  f32_to_f16_kernel<<<(MTOT * DIMC) / 256, 256, 0, stream>>>(x, xh, MTOT * DIMC);
  transpose_to_f16_kernel<<<(DIMC * 3 * DIMC) / 256, 256, 0, stream>>>(
      Wqkv, Wt1, DIMC, 3 * DIMC);
  transpose_to_f16_kernel<<<(DIMC * DIMC) / 256, 256, 0, stream>>>(
      Wproj, Wt2, DIMC, DIMC);
  gemm_qkv_kernel<<<dim3(24, 64), 256, 0, stream>>>(xh, Wt1, bqkv, Qh, Kh, Vt);
  flash_attn_kernel<<<dim3(SEQ / 64, NHEAD, BATCH), 128, 0, stream>>>(
      Qh, Kh, Vt, mask, temp, Oh);
  gemm_proj_kernel<<<dim3(8, 64), 256, 0, stream>>>(Oh, Wt2, bproj, out);
}
